// RNNLM_36369783063082
// MI455X (gfx1250) — matrix operand order repaired and checked by reading, not compiled, not run on hardware
//
#include <hip/hip_runtime.h>

typedef __attribute__((ext_vector_type(16))) _Float16 v16h;
typedef __attribute__((ext_vector_type(8)))  float    v8f;

#define VOCAB  32000
#define EMB    32
#define HID    16
#define SEQ    128
#define BATCH  32
#define ROWS   (SEQ * BATCH)     // 4096
#define RTILES (ROWS / 16)       // 256 row tiles of 16
#define CTILES (VOCAB / 16)      // 2000 vocab tiles of 16
#define NCHUNK 16                // vocab chunks
#define TPC    (CTILES / NCHUNK) // 125 tiles per chunk
#define WPB    4                 // waves (row tiles) per block

// ---- cross-lane xor reductions over 16-lane groups via ds_swizzle ---------
template <int MASK>
__device__ __forceinline__ float swz_xor(float x) {
    return __int_as_float(
        __builtin_amdgcn_ds_swizzle(__float_as_int(x), (MASK << 10) | 0x1F));
}
__device__ __forceinline__ float redmax16(float v) {
    v = fmaxf(v, swz_xor<1>(v));
    v = fmaxf(v, swz_xor<2>(v));
    v = fmaxf(v, swz_xor<4>(v));
    v = fmaxf(v, swz_xor<8>(v));
    return v;
}
__device__ __forceinline__ float redsum16(float v) {
    v += swz_xor<1>(v);
    v += swz_xor<2>(v);
    v += swz_xor<4>(v);
    v += swz_xor<8>(v);
    return v;
}

__device__ __forceinline__ v8f wmma_tile(v16h a, v16h b) {
    v8f d = {};
    return __builtin_amdgcn_wmma_f32_16x16x32_f16(false, a, false, b,
                                                  (short)0, d, false, false);
}

// ---------------------------------------------------------------------------
// Kernel 0: pack Wo [HID=16, VOCAB] f32 -> f16 B-fragment layout for
// v_wmma_f32_16x16x32_f16.  B is 32(K)x16(N): column N=lane%16; lanes 0-15
// hold K=0..7 in elements 0..7, lanes 16-31 hold K=8..15; elements 8..15 == 0 (pad).
// ---------------------------------------------------------------------------
__global__ void pack_b(const float* __restrict__ Wo, _Float16* __restrict__ Bp) {
    int gid = blockIdx.x * blockDim.x + threadIdx.x;   // 0 .. CTILES*32-1
    if (gid >= CTILES * 32) return;
    int tile = gid >> 5;
    int lane = gid & 31;
    v16h v;
    {
        int col = tile * 16 + (lane & 15);
        int kb = (lane < 16) ? 0 : 8;
#pragma unroll
        for (int k = 0; k < 8; ++k) v[k] = (_Float16)Wo[(kb + k) * VOCAB + col];
#pragma unroll
        for (int k = 8; k < 16; ++k) v[k] = (_Float16)0.0f;
    }
    *(v16h*)(Bp + (size_t)gid * 16) = v;
}

// ---------------------------------------------------------------------------
// Kernel 1: sequential tanh recurrence.  One block, thread = (b, h).
// Weight columns in registers; x and H staged in LDS.  Emits H_t directly in
// the WMMA A-fragment layout (M = lane%16; lanes 0-15 carry K=0..7, lanes
// 16-31 carry K=8..15, K>=16 zero-padded).
// ---------------------------------------------------------------------------
__global__ void rnn_h(const int* __restrict__ inp, const float* __restrict__ lookup,
                      const float* __restrict__ Wx, const float* __restrict__ Wh,
                      _Float16* __restrict__ Ap) {
    __shared__ float xs[BATCH * EMB];   // [b][e]
    __shared__ float Hs[BATCH * HID];   // [b][h]
    const int tid = threadIdx.x;        // 0..511
    const int b   = tid & 31;
    const int h   = tid >> 5;           // 0..15

    float wx[EMB], wh[HID];
#pragma unroll
    for (int e = 0; e < EMB; ++e) wx[e] = Wx[e * HID + h];
#pragma unroll
    for (int k = 0; k < HID; ++k) wh[k] = Wh[k * HID + h];

    Hs[b * HID + h] = 1.0f;             // reference inits H to ones
    __syncthreads();

    for (int t = 0; t < SEQ; ++t) {
        {   // stage x_t = lookup[input[t]] : 1024 floats, 2 per thread
            int i0 = tid, i1 = tid + 512;
            int b0 = i0 >> 5, e0 = i0 & 31;
            int b1 = i1 >> 5, e1 = i1 & 31;
            xs[i0] = lookup[(size_t)inp[t * BATCH + b0] * EMB + e0];
            xs[i1] = lookup[(size_t)inp[t * BATCH + b1] * EMB + e1];
        }
        __syncthreads();

        float acc = 0.0f;
#pragma unroll
        for (int e = 0; e < EMB; ++e) acc += xs[b * EMB + e] * wx[e];
#pragma unroll
        for (int k = 0; k < HID; ++k) acc += Hs[b * HID + k] * wh[k];
        float v = tanhf(acc);
        __syncthreads();                 // all reads of old H done

        Hs[b * HID + h] = v;

        int row = t * BATCH + b;
        int r = row >> 4, M = row & 15;
        _Float16 hv = (_Float16)v;
        if (h < 8) {
            size_t base = ((size_t)(r * 32 + M)) * 16;
            Ap[base + h]     = hv;               // K=h
            Ap[base + 8 + h] = (_Float16)0.0f;   // K=16..23 pad
        } else {
            size_t base = ((size_t)(r * 32 + 16 + M)) * 16;
            Ap[base + (h - 8)] = hv;             // K=h
            Ap[base + h]       = (_Float16)0.0f; // K=24..31 pad
        }
        __syncthreads();                 // new H visible before next step
    }
}

// ---------------------------------------------------------------------------
// Kernel 2: per-(rowtile, chunk) log-sum-exp partials via WMMA.
// Block = 4 waves sharing one vocab chunk (WGP cache reuse of B fragments).
// D layout: VGPR g, lanes 0-15 -> M=g, lanes 16-31 -> M=g+8; N = lane%16.
// B fragment loads are double-buffered against the WMMA.
// ---------------------------------------------------------------------------
__global__ void __launch_bounds__(WPB * 32)
lse_partial(const _Float16* __restrict__ Ap, const _Float16* __restrict__ Bp,
            float* __restrict__ part) {
    const int wave = threadIdx.x >> 5;
    const int lane = threadIdx.x & 31;
    const int r = blockIdx.x * WPB + wave;   // row tile
    const int c = blockIdx.y;                // vocab chunk
    const int t0 = c * TPC;

    v16h a = *(const v16h*)(Ap + ((size_t)(r * 32 + lane)) * 16);
    const _Float16* bbase = Bp + ((size_t)(t0 * 32 + lane)) * 16;

    // ---- pass A: row max -------------------------------------------------
    float m[8];
#pragma unroll
    for (int g = 0; g < 8; ++g) m[g] = -1e30f;
    {
        const _Float16* bp = bbase;
        v16h bf = *(const v16h*)bp;
        for (int i = 0; i < TPC - 1; ++i) {
            bp += 32 * 16;
            v16h bn = *(const v16h*)bp;          // prefetch next tile
            v8f d = wmma_tile(a, bf);
#pragma unroll
            for (int g = 0; g < 8; ++g) m[g] = (d[g] > m[g]) ? d[g] : m[g];
            bf = bn;
        }
        v8f d = wmma_tile(a, bf);
#pragma unroll
        for (int g = 0; g < 8; ++g) m[g] = (d[g] > m[g]) ? d[g] : m[g];
    }
#pragma unroll
    for (int g = 0; g < 8; ++g) m[g] = redmax16(m[g]);

    // ---- pass B: sum of exp(logit - rowmax) ------------------------------
    float s[8];
#pragma unroll
    for (int g = 0; g < 8; ++g) s[g] = 0.0f;
    {
        const _Float16* bp = bbase;
        v16h bf = *(const v16h*)bp;
        for (int i = 0; i < TPC - 1; ++i) {
            bp += 32 * 16;
            v16h bn = *(const v16h*)bp;
            v8f d = wmma_tile(a, bf);
#pragma unroll
            for (int g = 0; g < 8; ++g) s[g] += __expf(d[g] - m[g]);
            bf = bn;
        }
        v8f d = wmma_tile(a, bf);
#pragma unroll
        for (int g = 0; g < 8; ++g) s[g] += __expf(d[g] - m[g]);
    }
#pragma unroll
    for (int g = 0; g < 8; ++g) s[g] = redsum16(s[g]);

    if (lane == 0 || lane == 16) {
        int hi = (lane == 16) ? 8 : 0;
#pragma unroll
        for (int g = 0; g < 8; ++g) {
            int row = r * 16 + hi + g;
            part[((size_t)row * NCHUNK + c) * 2 + 0] = m[g];
            part[((size_t)row * NCHUNK + c) * 2 + 1] = s[g];
        }
    }
}

// ---------------------------------------------------------------------------
// Kernel 2b: combine chunk partials -> logZ[row] = M + log(sum s*exp(m-M))
// ---------------------------------------------------------------------------
__global__ void lse_combine(const float* __restrict__ part, float* __restrict__ logZ) {
    int row = blockIdx.x * blockDim.x + threadIdx.x;
    if (row >= ROWS) return;
    float m = -1e30f;
#pragma unroll
    for (int c = 0; c < NCHUNK; ++c) {
        float pm = part[((size_t)row * NCHUNK + c) * 2];
        m = (pm > m) ? pm : m;
    }
    float s = 0.0f;
#pragma unroll
    for (int c = 0; c < NCHUNK; ++c)
        s += part[((size_t)row * NCHUNK + c) * 2 + 1] *
             __expf(part[((size_t)row * NCHUNK + c) * 2] - m);
    logZ[row] = m + __logf(s);
}

// ---------------------------------------------------------------------------
// Kernel 3: recompute logits via WMMA, stream out = logit - logZ with
// non-temporal stores (512 MB written exactly once; keep Ap/Bp in L2).
// ---------------------------------------------------------------------------
__global__ void __launch_bounds__(WPB * 32)
write_out(const _Float16* __restrict__ Ap, const _Float16* __restrict__ Bp,
          const float* __restrict__ logZ, float* __restrict__ out) {
    const int wave = threadIdx.x >> 5;
    const int lane = threadIdx.x & 31;
    const int r = blockIdx.x * WPB + wave;
    const int c = blockIdx.y;
    const int t0 = c * TPC;

    v16h a = *(const v16h*)(Ap + ((size_t)(r * 32 + lane)) * 16);
    const int hi = (lane < 16) ? 0 : 8;
    const int nn = lane & 15;

    float lz[8];
    float* optr[8];
#pragma unroll
    for (int g = 0; g < 8; ++g) {
        int row = r * 16 + hi + g;
        lz[g] = logZ[row];
        optr[g] = out + (size_t)row * VOCAB + nn + (size_t)t0 * 16;
    }

    const _Float16* bp = Bp + ((size_t)(t0 * 32 + lane)) * 16;
    v16h bf = *(const v16h*)bp;
    for (int i = 0; i < TPC - 1; ++i) {
        bp += 32 * 16;
        v16h bn = *(const v16h*)bp;              // prefetch next tile
        v8f d = wmma_tile(a, bf);
#pragma unroll
        for (int g = 0; g < 8; ++g) {
            __builtin_nontemporal_store(d[g] - lz[g], optr[g]);
            optr[g] += 16;
        }
        bf = bn;
    }
    v8f d = wmma_tile(a, bf);
#pragma unroll
    for (int g = 0; g < 8; ++g)
        __builtin_nontemporal_store(d[g] - lz[g], optr[g]);
}

// ---------------------------------------------------------------------------
extern "C" void kernel_launch(void* const* d_in, const int* in_sizes, int n_in,
                              void* d_out, int out_size, void* d_ws, size_t ws_size,
                              hipStream_t stream) {
    const int*   inp    = (const int*)d_in[0];     // [SEQ, BATCH]
    const float* lookup = (const float*)d_in[1];   // [VOCAB, EMB]
    const float* Wx     = (const float*)d_in[2];   // [EMB, HID]
    const float* Wh     = (const float*)d_in[3];   // [HID, HID]
    const float* Wo     = (const float*)d_in[4];   // [HID, VOCAB]
    float*       out    = (float*)d_out;           // [SEQ, BATCH, VOCAB]

    char* ws = (char*)d_ws;
    _Float16* Ap   = (_Float16*)ws;                               // 256 KB
    _Float16* Bp   = (_Float16*)(ws + 262144);                    // 2.0 MB
    float*    part = (float*)(ws + 262144 + 2048000);             // 512 KB
    float*    logZ = (float*)(ws + 262144 + 2048000 + 524288);    // 16 KB

    pack_b<<<(CTILES * 32 + 255) / 256, 256, 0, stream>>>(Wo, Bp);
    rnn_h<<<1, 512, 0, stream>>>(inp, lookup, Wx, Wh, Ap);
    lse_partial<<<dim3(RTILES / WPB, NCHUNK), WPB * 32, 0, stream>>>(Ap, Bp, part);
    lse_combine<<<ROWS / 256, 256, 0, stream>>>(part, logZ);
    write_out<<<dim3(RTILES / WPB, NCHUNK), WPB * 32, 0, stream>>>(Ap, Bp, logZ, out);
}
